// MoE_29652454212575
// MI455X (gfx1250) — matrix operand order repaired and checked by reading, not compiled, not run on hardware
//
#include <hip/hip_runtime.h>
#include <math.h>

#define EMB   1024
#define HID   2048
#define NEXP  8
#define NTOK  32768      // 8 * 4096
#define CAP   5120       // int(1.25 * NTOK / NEXP)

typedef __attribute__((ext_vector_type(16))) __bf16 v16bf;
typedef __attribute__((ext_vector_type(8)))  __bf16 v8bf;
typedef __attribute__((ext_vector_type(4)))  __bf16 v4bf;
typedef __attribute__((ext_vector_type(8)))  float  v8f;

// ---- CDNA5 async global->LDS helpers -------------------------------------
// LDS flat addresses have the byte offset in the low 32 bits (ISA 10.2).
static __device__ __forceinline__ unsigned lds_lo32(const void* p) {
  return (unsigned)(unsigned long long)p;
}
static __device__ __forceinline__ void async_copy_b128(unsigned ldsAddr, const void* gaddr) {
  asm volatile("global_load_async_to_lds_b128 %0, %1, off"
               :: "v"(ldsAddr), "v"((unsigned long long)gaddr)
               : "memory");
}
static __device__ __forceinline__ void wait_async0() {
  asm volatile("s_wait_asynccnt 0" ::: "memory");
}

// ---- 1) Router: logits, z-loss partial, top-2 ----------------------------
// one wave (32 lanes) per token
__global__ __launch_bounds__(256) void router_k(
    const float* __restrict__ x, const float* __restrict__ wr,
    const float* __restrict__ br, int2* __restrict__ topk,
    float* __restrict__ zsum) {
  const int wave = threadIdx.x >> 5;
  const int lane = threadIdx.x & 31;
  const int tok  = blockIdx.x * 8 + wave;
  const float* xr = x + (size_t)tok * EMB;

  float p[NEXP];
#pragma unroll
  for (int e = 0; e < NEXP; ++e) p[e] = 0.f;

  for (int i = lane; i < EMB; i += 32) {
    float xv = xr[i];
    const float4* wrow = (const float4*)(wr + (size_t)i * NEXP);
    float4 w0 = wrow[0], w1 = wrow[1];
    p[0] += xv * w0.x; p[1] += xv * w0.y; p[2] += xv * w0.z; p[3] += xv * w0.w;
    p[4] += xv * w1.x; p[5] += xv * w1.y; p[6] += xv * w1.z; p[7] += xv * w1.w;
  }
#pragma unroll
  for (int e = 0; e < NEXP; ++e) {
    float v = p[e];
#pragma unroll
    for (int off = 16; off > 0; off >>= 1) v += __shfl_down(v, off, 32);
    p[e] = v;
  }
  if (lane == 0) {
    float ss = 0.f;
#pragma unroll
    for (int e = 0; e < NEXP; ++e) { p[e] += br[e]; ss += p[e] * p[e]; }
    // top-2 (ties -> lower index, matching lax.top_k)
    int i0 = 0; float m0 = p[0];
#pragma unroll
    for (int e = 1; e < NEXP; ++e) if (p[e] > m0) { m0 = p[e]; i0 = e; }
    int i1 = -1; float m1 = -3.4e38f;
#pragma unroll
    for (int e = 0; e < NEXP; ++e) if (e != i0 && p[e] > m1) { m1 = p[e]; i1 = e; }
    topk[tok] = make_int2(i0, i1);
    atomicAdd(zsum, ss);
  }
}

// ---- 2) Routing scan: cumsum + capacity, build buf[e][CAP] ---------------
__global__ void scan_k(const int2* __restrict__ topk, int* __restrict__ buf,
                       int* __restrict__ usage) {
  const int e = threadIdx.x >> 5;      // 8 waves, one per expert
  const int lane = threadIdx.x & 31;
  int* be = buf + e * CAP;
  int cnt = 0;
  for (int base = 0; base < NTOK; base += 32) {
    int n = base + lane;
    int2 t = topk[n];
    bool r = (t.x == e) || (t.y == e);
    unsigned m = __builtin_amdgcn_ballot_w32(r);
    int pos = cnt + __builtin_popcount(m & ((1u << lane) - 1u));
    if (r && pos < CAP) be[pos] = n;
    cnt += __builtin_popcount(m);
  }
  int used = cnt < CAP ? cnt : CAP;
  for (int c = used + lane; c < CAP; c += 32) be[c] = NTOK;   // pads
  if (lane == 0) usage[e] = used;
}

// ---- 3) Weight transpose + f32->bf16: w[e][R][C] -> wt[e][C][R] ----------
__global__ void wt_k(const float* __restrict__ w, __bf16* __restrict__ wt,
                     int R, int C) {
  size_t i = (size_t)blockIdx.x * 256 + threadIdx.x;   // over NEXP*R*C
  size_t RC = (size_t)R * C;
  int e = (int)(i / RC);
  size_t rem = i - (size_t)e * RC;
  int r = (int)(rem / C);
  int c = (int)(rem - (size_t)r * C);
  wt[(size_t)e * RC + (size_t)c * R + r] = (__bf16)w[i];
}

// ---- 4) Tiled bf16 WMMA GEMM: 128x128 tile, BK=64, 8 waves, dbl-buffer ---
// FFN1: A = gather(x[buf], fp32->bf16), B = w1t, epilogue = gelu -> h(bf16)
// FFN2: A = h (bf16, async),            B = w2t, epilogue = scatter-add out
template <int KD, int ND, bool ISF1>
__global__ __launch_bounds__(256) void ffn_k(
    const float*  __restrict__ x,      // FFN1 only
    const __bf16* __restrict__ hsrc,   // FFN2 only
    const __bf16* __restrict__ wt,     // [NEXP][ND][KD] bf16
    const float*  __restrict__ bias,   // [NEXP][ND]
    const int*    __restrict__ buf,    // [NEXP][CAP]
    __bf16*       __restrict__ hdst,   // FFN1 out
    float*        __restrict__ out)    // FFN2 out (atomic add)
{
  __shared__ __bf16 As[2][128 * 64];
  __shared__ __bf16 Bs[2][128 * 64];

  const int e   = blockIdx.z;
  const int m0  = blockIdx.x * 128;
  const int n0  = blockIdx.y * 128;
  const int tid = threadIdx.x;
  const int lane = tid & 31, wave = tid >> 5;
  const int wm = (wave >> 2) * 64;     // wave M offset: 0 / 64
  const int wn = (wave & 3) * 32;      // wave N offset: 0..96
  const int hs = (lane >> 4) & 1;
  const int ln = lane & 15;

  // init accumulators with bias (broadcast along rows; D lane holds col ln)
  v8f acc[4][2];
#pragma unroll
  for (int ni = 0; ni < 2; ++ni) {
    float b = bias[(size_t)e * ND + n0 + wn + ni * 16 + ln];
#pragma unroll
    for (int mi = 0; mi < 4; ++mi)
#pragma unroll
      for (int r = 0; r < 8; ++r) acc[mi][ni][r] = b;
  }

  // issue all global->LDS traffic for K-slice k0 into buffer `sel`
  auto load_tiles = [&](int k0, int sel) {
    const unsigned bL = lds_lo32(&Bs[sel][0]);
    // B tile [128 n][64 k] bf16 : async global->LDS, already n-major
#pragma unroll
    for (int i = 0; i < 4; ++i) {
      int c = tid + i * 256;            // 16-byte chunk id, 8 chunks/row
      int row = c >> 3;
      int col = (c & 7) * 16;           // byte offset in row (row = 128B)
      const char* g = (const char*)(wt + ((size_t)e * ND + n0 + row) * KD + k0) + col;
      async_copy_b128(bL + row * 128 + col, g);
    }
    if (ISF1) {
      // A tile: gather token rows (fp32) via buf, convert to bf16
      int row = tid >> 1;
      int cb  = (tid & 1) * 32;
      int tok = buf[e * CAP + m0 + row];
      __bf16* dst = &As[sel][0] + row * 64 + cb;
      if (tok < NTOK) {
        const float4* src = (const float4*)(x + (size_t)tok * EMB + k0 + cb);
#pragma unroll
        for (int i = 0; i < 8; ++i) {
          float4 v = src[i];
          *(v4bf*)(dst + i * 4) =
              (v4bf){(__bf16)v.x, (__bf16)v.y, (__bf16)v.z, (__bf16)v.w};
        }
      } else {
#pragma unroll
        for (int i = 0; i < 8; ++i)
          *(v4bf*)(dst + i * 4) = (v4bf){(__bf16)0.f, (__bf16)0.f, (__bf16)0.f, (__bf16)0.f};
      }
    } else {
      // A tile: h rows are contiguous bf16 -> async global->LDS
      const unsigned aL = lds_lo32(&As[sel][0]);
#pragma unroll
      for (int i = 0; i < 4; ++i) {
        int c = tid + i * 256;
        int row = c >> 3;
        int col = (c & 7) * 16;
        const char* g = (const char*)(hsrc + ((size_t)e * CAP + m0 + row) * KD + k0) + col;
        async_copy_b128(aL + row * 128 + col, g);
      }
    }
  };

  load_tiles(0, 0);                       // prologue: prefetch first slice

  for (int k0 = 0; k0 < KD; k0 += 64) {
    const int cur = (k0 >> 6) & 1;
    wait_async0();                        // this wave's copies for `cur` landed
    __syncthreads();                      // everyone landed + done reading cur^1
    if (k0 + 64 < KD) load_tiles(k0 + 64, cur ^ 1);   // overlap next DMA w/ math

    const __bf16* Ac = &As[cur][0];
    const __bf16* Bc = &Bs[cur][0];
#pragma unroll
    for (int kk = 0; kk < 64; kk += 32) {
      v16bf af[4], bfm[2];
#pragma unroll
      for (int mi = 0; mi < 4; ++mi) {
        // A frag: lane m holds K {0..7,16..23} (hs=0) or {8..15,24..31} (hs=1)
        const __bf16* p = Ac + (wm + mi * 16 + ln) * 64 + kk + hs * 8;
        v8bf lo = *(const v8bf*)p;
        v8bf hi = *(const v8bf*)(p + 16);
        af[mi] = __builtin_shufflevector(lo, hi, 0,1,2,3,4,5,6,7,8,9,10,11,12,13,14,15);
      }
#pragma unroll
      for (int ni = 0; ni < 2; ++ni) {
        // B frag: lane n holds K {0..7,16..23} (hs=0) or {8..15,24..31} (hs=1)
        const __bf16* p = Bc + (wn + ni * 16 + ln) * 64 + kk + hs * 8;
        v8bf lo = *(const v8bf*)p;
        v8bf hi = *(const v8bf*)(p + 16);
        bfm[ni] = __builtin_shufflevector(lo, hi, 0,1,2,3,4,5,6,7,8,9,10,11,12,13,14,15);
      }
#pragma unroll
      for (int mi = 0; mi < 4; ++mi)
#pragma unroll
        for (int ni = 0; ni < 2; ++ni)
          acc[mi][ni] = __builtin_amdgcn_wmma_f32_16x16x32_bf16(
              false, af[mi], false, bfm[ni], (short)0, acc[mi][ni], false, false);
    }
  }

  // epilogue: D lane (hs,ln), VGPR r -> element (row = hs*8 + r, col = ln)
#pragma unroll
  for (int mi = 0; mi < 4; ++mi) {
#pragma unroll
    for (int ni = 0; ni < 2; ++ni) {
#pragma unroll
      for (int r = 0; r < 8; ++r) {
        int mrow = m0 + wm + mi * 16 + hs * 8 + r;
        int ncol = n0 + wn + ni * 16 + ln;
        float v = acc[mi][ni][r];
        if (ISF1) {
          float g = 0.5f * v * (1.0f + erff(v * 0.70710678118654752f));  // exact gelu
          hdst[((size_t)e * CAP + mrow) * ND + ncol] = (__bf16)g;
        } else {
          int tok = buf[e * CAP + mrow];
          if (tok < NTOK) atomicAdd(out + (size_t)tok * ND + ncol, v);
        }
      }
    }
  }
}

// ---- 5) Final losses + usage --------------------------------------------
__global__ void loss_k(const float* __restrict__ zsum, const int* __restrict__ usage,
                       float* __restrict__ tail) {
  if (threadIdx.x == 0) {
    float z = 0.001f * zsum[0] / (float)((size_t)NTOK * NEXP);
    float u[NEXP], tot = 0.f;
#pragma unroll
    for (int e = 0; e < NEXP; ++e) { u[e] = (float)usage[e]; tot += u[e]; }
    float bal = 0.f;
#pragma unroll
    for (int e = 0; e < NEXP; ++e) { float l = u[e] / (tot + 1e-6f); bal += l * l; }
    tail[0] = (float)NEXP * bal + z;
#pragma unroll
    for (int e = 0; e < NEXP; ++e) tail[1 + e] = u[e];
  }
}

// ---- workspace layout (bytes) --------------------------------------------
static const size_t OFF_ZS  = 0;
static const size_t OFF_US  = 256;
static const size_t OFF_TK  = 512;
static const size_t OFF_BUF = OFF_TK  + (size_t)NTOK * 8;
static const size_t OFF_W1T = OFF_BUF + (size_t)NEXP * CAP * 4;
static const size_t OFF_W2T = OFF_W1T + (size_t)NEXP * EMB * HID * 2;
static const size_t OFF_H   = OFF_W2T + (size_t)NEXP * HID * EMB * 2;

extern "C" void kernel_launch(void* const* d_in, const int* in_sizes, int n_in,
                              void* d_out, int out_size, void* d_ws, size_t ws_size,
                              hipStream_t stream) {
  (void)in_sizes; (void)n_in; (void)ws_size;
  const float* x  = (const float*)d_in[0];
  const float* wr = (const float*)d_in[1];
  const float* br = (const float*)d_in[2];
  const float* w1 = (const float*)d_in[3];
  const float* b1 = (const float*)d_in[4];
  const float* w2 = (const float*)d_in[5];
  const float* b2 = (const float*)d_in[6];

  char* ws = (char*)d_ws;
  float* zsum  = (float*)(ws + OFF_ZS);
  int*   usage = (int*)  (ws + OFF_US);
  int2*  topk  = (int2*) (ws + OFF_TK);
  int*   buf   = (int*)  (ws + OFF_BUF);
  __bf16* w1t  = (__bf16*)(ws + OFF_W1T);
  __bf16* w2t  = (__bf16*)(ws + OFF_W2T);
  __bf16* hbuf = (__bf16*)(ws + OFF_H);
  float* out   = (float*)d_out;

  hipMemsetAsync(d_out, 0, (size_t)out_size * sizeof(float), stream);
  hipMemsetAsync(zsum, 0, sizeof(float), stream);

  router_k<<<NTOK / 8, 256, 0, stream>>>(x, wr, br, topk, zsum);
  scan_k<<<1, 256, 0, stream>>>(topk, buf, usage);
  wt_k<<<(NEXP * EMB * HID) / 256, 256, 0, stream>>>(w1, w1t, EMB, HID);
  wt_k<<<(NEXP * HID * EMB) / 256, 256, 0, stream>>>(w2, w2t, HID, EMB);

  ffn_k<EMB, HID, true><<<dim3(CAP / 128, HID / 128, NEXP), 256, 0, stream>>>(
      x, nullptr, w1t, b1, buf, hbuf, nullptr);
  ffn_k<HID, EMB, false><<<dim3(CAP / 128, EMB / 128, NEXP), 256, 0, stream>>>(
      nullptr, hbuf, w2t, b2, buf, nullptr, out);

  loss_k<<<1, 32, 0, stream>>>(zsum, usage, out + (size_t)NTOK * EMB);
}
